// AgnosticEmbeddedOneBodyVariableUpdate_86835648790775
// MI455X (gfx1250) — compile-verified, matrix operand order checked by reading, not run on hardware
//
#include <hip/hip_runtime.h>
#include <hip/hip_bf16.h>
#include <math.h>

#define N_NODES   25000
#define N_EDGES   500000
#define CH        128
#define SH_DIM    16
#define RADIAL_IN 8
#define RH        64
#define N_PATHS   4
#define POT_DIM   16
#define OUT_COLS  2048
#define AVG_NEIGH 20.0f
#define FIELD_NORM 10.0f

typedef __attribute__((ext_vector_type(16))) _Float16 v16h;
typedef __attribute__((ext_vector_type(8)))  _Float16 h8;
typedef __attribute__((ext_vector_type(8)))  float    v8f;

// ---- WMMA fragment helpers (CDNA5 16x16x32 f16 layouts, ISA 7.12.2) ----
// A (16x32 f16): lane L holds row M=L&15; its 16 halves are two contiguous
// 16-byte runs in a row-major LDS tile => two ds_load_b128.
__device__ __forceinline__ v16h load_a_frag(const _Float16* A, int ld, int k0, int lane) {
  int row = lane & 15;
  int koff = ((lane >> 4) & 1) << 3;
  const _Float16* p = A + row * ld + k0 + koff;
  h8 lo = *(const h8*)p;
  h8 hi = *(const h8*)(p + 16);
  return __builtin_shufflevector(lo, hi, 0, 1, 2, 3, 4, 5, 6, 7,
                                 8, 9, 10, 11, 12, 13, 14, 15);
}

// B fragment from weights pre-swizzled in LDS to fragment-major order:
//   swz[((kc*N + col)*2 + half)*16 + run*8 + j]  (contiguous 32B per lane)
__device__ __forceinline__ v16h load_b_frag_swz(const _Float16* Bswz, int ncols,
                                                int kc, int col0, int lane) {
  int col = col0 + (lane & 15);
  int half = (lane >> 4) & 1;
  const _Float16* p = Bswz + (size_t)(((kc * ncols + col) << 1) + half) * 16;
  h8 lo = *(const h8*)p;
  h8 hi = *(const h8*)(p + 8);
  return __builtin_shufflevector(lo, hi, 0, 1, 2, 3, 4, 5, 6, 7,
                                 8, 9, 10, 11, 12, 13, 14, 15);
}

// Stage a KxN f32 weight (K padded to Kpad, zeros beyond Ksrc, scaled) from
// global into fragment-major swizzled f16 LDS. Coalesced global reads.
__device__ __forceinline__ void stage_b_swz(_Float16* dst, const float* src,
                                            int Ksrc, int Kpad, int N, float scale,
                                            int tid, int stride) {
  for (int i = tid; i < Kpad * N; i += stride) {
    int k = i / N, n = i % N;
    float v = (k < Ksrc) ? src[k * N + n] * scale : 0.0f;
    int kc = k >> 5, kl = k & 31;
    int idx = ((((kc * N + n) << 1) + ((kl >> 3) & 1)) << 4) + ((kl >> 4) << 3) + (kl & 7);
    dst[idx] = (_Float16)v;
  }
}

__device__ __forceinline__ v8f wmma_f16(v16h a, v16h b, v8f c) {
  return __builtin_amdgcn_wmma_f32_16x16x32_f16(false, a, false, b, (short)0, c, false, false);
}

__device__ __forceinline__ v8f zero8() {
  v8f z = {0.f, 0.f, 0.f, 0.f, 0.f, 0.f, 0.f, 0.f};
  return z;
}

__device__ __forceinline__ float silu(float x) { return x / (1.0f + __expf(-x)); }

// Packed f16 atomic add (GLOBAL_ATOMIC_PK_ADD_F16, ISA 15.18.3 op 89).
__device__ __forceinline__ void atomic_pk_add_f16(_Float16* addr, float lo, float hi) {
  union { _Float16 f[2]; unsigned int u; } pk;
  pk.f[0] = (_Float16)lo;
  pk.f[1] = (_Float16)hi;
  asm volatile("global_atomic_pk_add_f16 %0, %1, off"
               :: "v"(addr), "v"(pk.u) : "memory");
}

// -------------------------------------------------------------------------
__global__ void k_zero(float* __restrict__ p, size_t n) {
  size_t i = (size_t)blockIdx.x * blockDim.x + threadIdx.x;
  size_t st = (size_t)gridDim.x * blockDim.x;
  for (; i < n; i += st) p[i] = 0.0f;
}

// -------------------------------------------------------------------------
// h = nf @ W_up ; one_body = (pf/10) @ Wp + nf @ Wn + q (x) Wc
// 128 threads = 4 waves, 64 nodes per block (16 per wave).
__global__ void k_node_linear(const float* __restrict__ nf, const float* __restrict__ pf,
                              const float* __restrict__ q,
                              const float* __restrict__ Wup, const float* __restrict__ Wn,
                              const float* __restrict__ Wp,  const float* __restrict__ Wc,
                              float* __restrict__ h_ws, float* __restrict__ ob_ws) {
  extern __shared__ char smem[];
  _Float16* A  = (_Float16*)smem;   // 64 x 128 row-major (node_feats tile)
  _Float16* W  = A + 64 * CH;       // 128x128 swizzled (W_up then Wn)
  _Float16* P  = W + CH * CH;       // 64 x 32 row-major (potential, K pad 16->32)
  _Float16* WP = P + 64 * 32;       // 32 x 128 swizzled (Wp, K padded)
  const int tid = threadIdx.x, lane = tid & 31, wave = tid >> 5;
  const int node0 = blockIdx.x * 64;

  for (int i = tid; i < 64 * CH; i += blockDim.x) {
    int r = i >> 7, c = i & (CH - 1);
    int n = node0 + r;
    A[i] = (n < N_NODES) ? (_Float16)nf[n * CH + c] : (_Float16)0.0f;
  }
  for (int i = tid; i < 64 * 32; i += blockDim.x) {
    int r = i >> 5, c = i & 31;
    int n = node0 + r;
    float v = (c < POT_DIM && n < N_NODES) ? pf[n * POT_DIM + c] * (1.0f / FIELD_NORM) : 0.0f;
    P[i] = (_Float16)v;
  }
  stage_b_swz(W, Wup, CH, CH, CH, 1.0f, tid, blockDim.x);
  stage_b_swz(WP, Wp, POT_DIM, 32, CH, 1.0f, tid, blockDim.x);
  __syncthreads();

  const _Float16* Aw = A + wave * 16 * CH;
  const _Float16* Pw = P + wave * 16 * 32;
  const int row0 = node0 + wave * 16;
  const int rb = (lane >> 4) << 3, n0l = lane & 15;

  v8f acc[8];
#pragma unroll
  for (int nt = 0; nt < 8; ++nt) acc[nt] = zero8();
#pragma unroll
  for (int kc = 0; kc < 4; ++kc) {
    v16h a = load_a_frag(Aw, CH, kc * 32, lane);
#pragma unroll
    for (int nt = 0; nt < 8; ++nt) {
      v16h b = load_b_frag_swz(W, CH, kc, nt * 16, lane);
      acc[nt] = wmma_f16(a, b, acc[nt]);
    }
  }
#pragma unroll
  for (int nt = 0; nt < 8; ++nt) {
    int col = nt * 16 + n0l;
#pragma unroll
    for (int r = 0; r < 8; ++r) {
      int row = row0 + rb + r;
      if (row < N_NODES) h_ws[(size_t)row * CH + col] = acc[nt][r];
    }
  }

  __syncthreads();
  stage_b_swz(W, Wn, CH, CH, CH, 1.0f, tid, blockDim.x);
  __syncthreads();

#pragma unroll
  for (int nt = 0; nt < 8; ++nt) acc[nt] = zero8();
#pragma unroll
  for (int kc = 0; kc < 4; ++kc) {
    v16h a = load_a_frag(Aw, CH, kc * 32, lane);
#pragma unroll
    for (int nt = 0; nt < 8; ++nt) {
      v16h b = load_b_frag_swz(W, CH, kc, nt * 16, lane);
      acc[nt] = wmma_f16(a, b, acc[nt]);
    }
  }
  {
    v16h ap = load_a_frag(Pw, 32, 0, lane);
#pragma unroll
    for (int nt = 0; nt < 8; ++nt) {
      v16h b = load_b_frag_swz(WP, CH, 0, nt * 16, lane);
      acc[nt] = wmma_f16(ap, b, acc[nt]);
    }
  }
#pragma unroll
  for (int nt = 0; nt < 8; ++nt) {
    int col = nt * 16 + n0l;
    float wc = Wc[col];
#pragma unroll
    for (int r = 0; r < 8; ++r) {
      int row = row0 + rb + r;
      if (row < N_NODES) ob_ws[(size_t)row * CH + col] = acc[nt][r] + q[row] * wc;
    }
  }
}

// -------------------------------------------------------------------------
// Fused edge kernel: radial MLP (WMMA) + gather h[sender] + scatter-add of
// m = hs * r * sh into f16 agg[recv][16][128] via packed f16 atomics.
// 256 threads = 8 waves, 16 edges per wave (128 per block).
__global__ void k_edge(const float* __restrict__ ef, const float* __restrict__ ea,
                       const int* __restrict__ eidx,
                       const float* __restrict__ W1, const float* __restrict__ W2,
                       const float* __restrict__ W3, const float* __restrict__ W4,
                       const float* __restrict__ h_ws, _Float16* __restrict__ agg) {
  extern __shared__ char smem[];
  _Float16* w1 = (_Float16*)smem;        // 32 x 64 swizzled (rows 8..31 zero)
  _Float16* w2 = w1 + 32 * RH;           // 64 x 64 swizzled
  _Float16* w3 = w2 + RH * RH;           // 64 x 64 swizzled
  _Float16* w4 = w3 + RH * RH;           // 64 x 512 swizzled
  _Float16* xa = w4 + RH * 512;          // 8 waves x (16 x 64) activation ping
  _Float16* xb = xa + 8 * 16 * RH;       // 8 waves x (16 x 64) activation pong
  float*   hsb = (float*)(xb + 8 * 16 * RH);    // 8 x (16 x 128) gathered h
  float*   shb = hsb + 8 * 16 * CH;             // 8 x (16 x 16) sph. harmonics
  int*     rcv = (int*)(shb + 8 * 16 * SH_DIM); // 8 x 16 receiver ids (-1 invalid)

  const int tid = threadIdx.x, lane = tid & 31, wave = tid >> 5;
  const int e0 = blockIdx.x * 128 + wave * 16;

  stage_b_swz(w1, W1, RADIAL_IN, 32, RH, 1.0f, tid, blockDim.x);
  stage_b_swz(w2, W2, RH, RH, RH, 1.0f, tid, blockDim.x);
  stage_b_swz(w3, W3, RH, RH, RH, 1.0f, tid, blockDim.x);
  stage_b_swz(w4, W4, RH, RH, 512, 1.0f, tid, blockDim.x);

  _Float16* xaw = xa + wave * 16 * RH;
  _Float16* xbw = xb + wave * 16 * RH;
  float* hsw = hsb + wave * 16 * CH;
  float* shw = shb + wave * 16 * SH_DIM;
  int* rcw = rcv + wave * 16;

  for (int i = lane; i < 16 * RH; i += 32) {   // edge_feats, K padded 8->64 zeros
    int r = i >> 6, c = i & 63;
    int e = e0 + r;
    float v = (c < RADIAL_IN && e < N_EDGES) ? ef[(size_t)e * RADIAL_IN + c] : 0.0f;
    xaw[i] = (_Float16)v;
  }
  for (int i = lane; i < 16 * SH_DIM; i += 32) {
    int r = i >> 4, c = i & 15;
    int e = e0 + r;
    shw[i] = (e < N_EDGES) ? ea[(size_t)e * SH_DIM + c] : 0.0f;
  }
  if (lane < 16) {
    int e = e0 + lane;
    rcw[lane] = (e < N_EDGES) ? eidx[N_EDGES + e] : -1;
  }
  for (int i = lane; i < 16 * CH; i += 32) {   // gather h[sender]
    int r = i >> 7, c = i & (CH - 1);
    int e = e0 + r;
    int s = (e < N_EDGES) ? eidx[e] : 0;
    hsw[i] = (e < N_EDGES) ? h_ws[(size_t)s * CH + c] : 0.0f;
  }
  __syncthreads();

  const int rb = (lane >> 4) << 3, n0l = lane & 15;

  // layer 1: silu(ef @ W1), K=32 (padded), N=64
  {
    v16h a = load_a_frag(xaw, RH, 0, lane);
    v8f acc[4];
#pragma unroll
    for (int nt = 0; nt < 4; ++nt) {
      v16h b = load_b_frag_swz(w1, RH, 0, nt * 16, lane);
      acc[nt] = wmma_f16(a, b, zero8());
    }
#pragma unroll
    for (int nt = 0; nt < 4; ++nt)
#pragma unroll
      for (int r = 0; r < 8; ++r)
        xbw[(rb + r) * RH + nt * 16 + n0l] = (_Float16)silu(acc[nt][r]);
  }
  // layer 2: silu(x @ W2)
  {
    v8f acc[4];
#pragma unroll
    for (int nt = 0; nt < 4; ++nt) acc[nt] = zero8();
#pragma unroll
    for (int kc = 0; kc < 2; ++kc) {
      v16h a = load_a_frag(xbw, RH, kc * 32, lane);
#pragma unroll
      for (int nt = 0; nt < 4; ++nt) {
        v16h b = load_b_frag_swz(w2, RH, kc, nt * 16, lane);
        acc[nt] = wmma_f16(a, b, acc[nt]);
      }
    }
#pragma unroll
    for (int nt = 0; nt < 4; ++nt)
#pragma unroll
      for (int r = 0; r < 8; ++r)
        xaw[(rb + r) * RH + nt * 16 + n0l] = (_Float16)silu(acc[nt][r]);
  }
  // layer 3: silu(x @ W3)
  {
    v8f acc[4];
#pragma unroll
    for (int nt = 0; nt < 4; ++nt) acc[nt] = zero8();
#pragma unroll
    for (int kc = 0; kc < 2; ++kc) {
      v16h a = load_a_frag(xaw, RH, kc * 32, lane);
#pragma unroll
      for (int nt = 0; nt < 4; ++nt) {
        v16h b = load_b_frag_swz(w3, RH, kc, nt * 16, lane);
        acc[nt] = wmma_f16(a, b, acc[nt]);
      }
    }
#pragma unroll
    for (int nt = 0; nt < 4; ++nt)
#pragma unroll
      for (int r = 0; r < 8; ++r)
        xbw[(rb + r) * RH + nt * 16 + n0l] = (_Float16)silu(acc[nt][r]);
  }
  // layer 4 (x @ W4, N=512) fused with message formation + packed scatter.
  // Lane pair (2k, 2k+1) holds channels (ch, ch+1): exchange via shfl_xor,
  // even lane issues one global_atomic_pk_add_f16 for both channels.
  {
    v16h a0 = load_a_frag(xbw, RH, 0, lane);
    v16h a1 = load_a_frag(xbw, RH, 32, lane);
    const int ldims[4] = {1, 3, 5, 7};
    const int loffs[4] = {0, 1, 4, 9};
    for (int nt = 0; nt < 32; ++nt) {
      v8f c = zero8();
      v16h b0 = load_b_frag_swz(w4, 512, 0, nt * 16, lane);
      c = wmma_f16(a0, b0, c);
      v16h b1 = load_b_frag_swz(w4, 512, 1, nt * 16, lane);
      c = wmma_f16(a1, b1, c);
      int p = nt >> 3;                       // path
      int ch = ((nt & 7) << 4) + n0l;        // channel 0..127
      int ld = ldims[p], lo = loffs[p];
#pragma unroll
      for (int r = 0; r < 8; ++r) {
        int el = rb + r;
        int node = rcw[el];
        float base = hsw[el * CH + ch] * c[r];
        float other = __shfl_xor(base, 1, 32);   // neighbor channel's value
        if (((lane & 1) == 0) && node >= 0) {
          _Float16* dst = agg + (size_t)node * (SH_DIM * CH) + (size_t)lo * CH + ch;
          for (int m = 0; m < ld; ++m) {
            float s = shw[el * SH_DIM + lo + m];
            atomic_pk_add_f16(dst + (size_t)m * CH, base * s, other * s);
          }
        }
      }
    }
  }
}

// -------------------------------------------------------------------------
// out[n, colbase[p] + d*L + m] = sum_c agg_f16[n][mg][c] * (W_out[p][c][d]/20)
// (1/AVG_NEIGH folded into the f16 W_out staging.)
// 128 threads = 4 waves, 64 nodes per block; wave w owns 16 nodes, all 16 mg.
__global__ void k_out(const _Float16* __restrict__ agg, const float* __restrict__ Wout,
                      const float* __restrict__ ob_ws, float* __restrict__ out) {
  extern __shared__ char smem[];
  _Float16* wo = (_Float16*)smem;            // 4 x 128x128 swizzled W_out/20
  _Float16* aT = wo + N_PATHS * CH * CH;     // 4 waves x (16 x 128) row-major
  float* obl = (float*)(aT + 4 * 16 * CH);   // 64 x 128 one_body tile
  const int tid = threadIdx.x, lane = tid & 31, wave = tid >> 5;
  const int node0 = blockIdx.x * 64;
  const int nodew = node0 + wave * 16;

#pragma unroll
  for (int p = 0; p < N_PATHS; ++p)
    stage_b_swz(wo + p * CH * CH, Wout + (size_t)p * CH * CH, CH, CH, CH,
                1.0f / AVG_NEIGH, tid, blockDim.x);
  for (int i = tid; i < 64 * CH; i += blockDim.x) {
    int r = i >> 7, c = i & (CH - 1);
    int n = node0 + r;
    obl[i] = (n < N_NODES) ? ob_ws[(size_t)n * CH + c] : 0.0f;
  }
  __syncthreads();

  const int path_of[16] = {0, 1, 1, 1, 2, 2, 2, 2, 2, 3, 3, 3, 3, 3, 3, 3};
  const int ldims[4]   = {1, 3, 5, 7};
  const int loffs[4]   = {0, 1, 4, 9};
  const int colbase[4] = {0, 128, 512, 1152};

  _Float16* aw = aT + wave * 16 * CH;
  const float* oblw = obl + wave * 16 * CH;
  const int rb = (lane >> 4) << 3, n0l = lane & 15;

  for (int mg = 0; mg < 16; ++mg) {
    int p = path_of[mg];
    int mloc = mg - loffs[p];
    // stage this wave's 16x128 f16 A tile (pure copy, 32-bit moves)
    for (int i = lane; i < 16 * (CH / 2); i += 32) {
      int r = i >> 6, c = i & 63;
      int n = nodew + r;
      unsigned int v = (n < N_NODES)
          ? ((const unsigned int*)(agg + (size_t)n * (SH_DIM * CH) + (size_t)mg * CH))[c]
          : 0u;
      ((unsigned int*)aw)[i] = v;
    }
    // same-wave LDS ops are in-order: safe to read back immediately
    v16h af[4];
#pragma unroll
    for (int kc = 0; kc < 4; ++kc) af[kc] = load_a_frag(aw, CH, kc * 32, lane);
    const _Float16* wop = wo + p * CH * CH;
    int ld = ldims[p], cb = colbase[p];
#pragma unroll
    for (int nt = 0; nt < 8; ++nt) {
      v8f c = zero8();
#pragma unroll
      for (int kc = 0; kc < 4; ++kc) {
        v16h b = load_b_frag_swz(wop, CH, kc, nt * 16, lane);
        c = wmma_f16(af[kc], b, c);
      }
      int d = nt * 16 + n0l;
#pragma unroll
      for (int r = 0; r < 8; ++r) {
        int n = nodew + rb + r;
        if (n < N_NODES) {
          float v = c[r];
          if (p == 0) v += oblw[(rb + r) * CH + d];
          out[(size_t)n * OUT_COLS + cb + d * ld + mloc] = v;
        }
      }
    }
  }
}

// -------------------------------------------------------------------------
extern "C" void kernel_launch(void* const* d_in, const int* in_sizes, int n_in,
                              void* d_out, int out_size, void* d_ws, size_t ws_size,
                              hipStream_t stream) {
  (void)in_sizes; (void)n_in; (void)out_size; (void)ws_size;
  const float* node_feats = (const float*)d_in[1];
  const float* edge_attrs = (const float*)d_in[2];
  const float* edge_feats = (const float*)d_in[3];
  const int*   edge_index = (const int*)d_in[4];
  const float* pot        = (const float*)d_in[5];
  const float* q          = (const float*)d_in[6];
  const float* Wup        = (const float*)d_in[7];
  const float* W1         = (const float*)d_in[8];
  const float* W2         = (const float*)d_in[9];
  const float* W3         = (const float*)d_in[10];
  const float* W4         = (const float*)d_in[11];
  const float* Wout       = (const float*)d_in[12];
  const float* Wp         = (const float*)d_in[13];
  const float* Wn         = (const float*)d_in[14];
  const float* Wc         = (const float*)d_in[15];
  float* out = (float*)d_out;

  // workspace: h f32 (12.8MB) | one_body f32 (12.8MB) | agg f16 (102.4MB,
  // fully L2-resident on MI455X's 192MB L2)
  float* h_ws  = (float*)d_ws;
  float* ob_ws = h_ws + (size_t)N_NODES * CH;
  _Float16* agg = (_Float16*)(ob_ws + (size_t)N_NODES * CH);

  k_zero<<<2048, 256, 0, stream>>>((float*)agg, (size_t)N_NODES * SH_DIM * CH / 2);

  size_t lds1 = (size_t)(64 * CH + CH * CH + 64 * 32 + 32 * CH) * sizeof(_Float16);
  k_node_linear<<<(N_NODES + 63) / 64, 128, lds1, stream>>>(
      node_feats, pot, q, Wup, Wn, Wp, Wc, h_ws, ob_ws);

  size_t lds2 = (size_t)(32 * RH + RH * RH + RH * RH + RH * 512 + 16 * 16 * RH) * sizeof(_Float16)
              + (size_t)(8 * 16 * CH + 8 * 16 * SH_DIM) * sizeof(float)
              + (size_t)(8 * 16) * sizeof(int);
  k_edge<<<(N_EDGES + 127) / 128, 256, lds2, stream>>>(
      edge_feats, edge_attrs, edge_index, W1, W2, W3, W4, h_ws, agg);

  size_t lds3 = (size_t)(N_PATHS * CH * CH + 4 * 16 * CH) * sizeof(_Float16)
              + (size_t)(64 * CH) * sizeof(float);
  k_out<<<(N_NODES + 63) / 64, 128, lds3, stream>>>(agg, Wout, ob_ws, out);
}
